// BinaryBenchMarkNet_MutiQ_8392366096949
// MI455X (gfx1250) — matrix operand order repaired and checked by reading, not compiled, not run on hardware
//
#include <hip/hip_runtime.h>

// ---------------- problem constants ----------------
#define BB   16384
#define DD   512
#define HH1  1024
#define HH2  2048
#define KK   11
#define EPSV 1e-5f
#define THRV 0.25f
#define SPART 64   // row partitions for column stats

// ---------------- WMMA types ----------------
typedef __attribute__((ext_vector_type(16))) __bf16 v16bf;
typedef __attribute__((ext_vector_type(8)))  __bf16 v8bf;
typedef __attribute__((ext_vector_type(8)))  float  v8f;

// ---------------- fp32 -> bf16 (RNE) ----------------
__device__ __forceinline__ unsigned short f2bf(float x) {
  unsigned int u = __float_as_uint(x);
  unsigned int r = (u + 0x7fffu + ((u >> 16) & 1u)) >> 16;
  return (unsigned short)r;
}

__global__ __launch_bounds__(256) void cvt_bf16_kernel(const float* __restrict__ in,
                                                       unsigned short* __restrict__ out,
                                                       size_t n) {
  size_t i = (size_t)blockIdx.x * blockDim.x + threadIdx.x;
  size_t stride = (size_t)gridDim.x * blockDim.x;
  for (; i < n; i += stride) out[i] = f2bf(in[i]);
}

// ---------------- fp32 [Kd][N] -> bf16 transposed [N][Kd] (batched over z) ----
// 32x32 tiles via LDS; Kd%32==0, N%32==0.
__global__ __launch_bounds__(256) void cvt_t_kernel(const float* __restrict__ in,
                                                    unsigned short* __restrict__ out,
                                                    int Kd, int N, size_t slice) {
  __shared__ float tile[32][33];
  const float* src = in + (size_t)blockIdx.z * slice;
  unsigned short* dst = out + (size_t)blockIdx.z * slice;
  const int bk = blockIdx.y * 32;
  const int bn = blockIdx.x * 32;
  const int tx = threadIdx.x & 31;
  const int ty = threadIdx.x >> 5;  // 0..7
#pragma unroll
  for (int i = 0; i < 4; ++i) {
    int k = ty + i * 8;
    tile[k][tx] = src[(size_t)(bk + k) * N + bn + tx];
  }
  __syncthreads();
#pragma unroll
  for (int i = 0; i < 4; ++i) {
    int n = ty + i * 8;
    dst[(size_t)(bn + n) * Kd + bk + tx] = f2bf(tile[tx][n]);
  }
}

// ---------------- bf16 WMMA GEMM: C[M,N] = A[M,Kd] @ B[Kd,N] ----------------
// A row-major bf16 [M][Kd]; BT = B transposed, bf16 [N][Kd]; C fp32 [M][N].
// M%128==0, N%128==0, Kd%32==0. 256 threads = 8 waves; 128x128 block tile;
// double-buffered LDS; each wave: 32x64 slab = 2x4 accs = 8 WMMAs per k-step.
#define BM 128
#define BN 128
#define BK 32
#define LSTR 40  // padded LDS row stride (elems): conflict-free frag reads

__global__ __launch_bounds__(256) void gemm_bf16_wmma(
    const unsigned short* __restrict__ A, const unsigned short* __restrict__ BT,
    float* __restrict__ C, int M, int N, int Kd) {
  __shared__ __align__(32) unsigned short sA[2][BM * LSTR];
  __shared__ __align__(32) unsigned short sB[2][BN * LSTR];

  const int tid  = threadIdx.x;
  const int wave = tid >> 5;
  const int lane = tid & 31;
  const int rowBlk = blockIdx.y * BM;
  const int colBlk = blockIdx.x * BN;
  const int waveM = (wave & 3) * 32;   // 4 M groups of 32
  const int waveN = (wave >> 2) * 64;  // 2 N groups of 64

  const int lr     = lane & 15;
  const int aKhalf = (lane >> 4) * 8;   // A frag: lanes<16 hold K 0..7 & 16..23
  const int bK0    = (lane >> 4) * 8;   // B frag: lanes<16 hold K 0..7 & 16..23

  // global load pointers: two 8-elem chunks per thread per tile (128x32 tiles)
  const int r0 = tid >> 2;            // 0..63
  const int c8 = (tid & 3) * 8;       // 0,8,16,24
  const unsigned short* aP0 = A  + (size_t)(rowBlk + r0) * Kd + c8;
  const unsigned short* aP1 = aP0 + (size_t)64 * Kd;
  const unsigned short* bP0 = BT + (size_t)(colBlk + r0) * Kd + c8;
  const unsigned short* bP1 = bP0 + (size_t)64 * Kd;

  unsigned short* sA0 = &sA[0][r0 * LSTR + c8];
  unsigned short* sB0 = &sB[0][r0 * LSTR + c8];
  const int bufStep = BM * LSTR;  // elems per buffer

  v8f acc[2][4];
#pragma unroll
  for (int i = 0; i < 2; ++i)
#pragma unroll
    for (int j = 0; j < 4; ++j) acc[i][j] = (v8f){};

  // prologue: stage 0 -> buffer 0
  uint4 a0 = *(const uint4*)aP0;
  uint4 a1 = *(const uint4*)aP1;
  uint4 b0 = *(const uint4*)bP0;
  uint4 b1 = *(const uint4*)bP1;
  aP0 += BK; aP1 += BK; bP0 += BK; bP1 += BK;
  *(uint4*)(sA0)                 = a0;
  *(uint4*)(sA0 + 64 * LSTR)     = a1;
  *(uint4*)(sB0)                 = b0;
  *(uint4*)(sB0 + 64 * LSTR)     = b1;
  __syncthreads();

  const int nStages = Kd / BK;
  for (int s = 0; s < nStages - 1; ++s) {
    const int cur = s & 1;
    const int nxt = cur ^ 1;
    // prefetch next stage from global
    a0 = *(const uint4*)aP0;
    a1 = *(const uint4*)aP1;
    b0 = *(const uint4*)bP0;
    b1 = *(const uint4*)bP1;
    aP0 += BK; aP1 += BK; bP0 += BK; bP1 += BK;

    // compute current stage
    {
      union F { v16bf v; v8bf h[2]; };
      F afr[2], bfr[4];
#pragma unroll
      for (int i = 0; i < 2; ++i) {
        const unsigned short* p = &sA[cur][(waveM + 16 * i + lr) * LSTR];
        afr[i].h[0] = *(const v8bf*)(p + aKhalf);
        afr[i].h[1] = *(const v8bf*)(p + aKhalf + 16);
      }
#pragma unroll
      for (int j = 0; j < 4; ++j) {
        const unsigned short* p = &sB[cur][(waveN + 16 * j + lr) * LSTR + bK0];
        bfr[j].h[0] = *(const v8bf*)(p);
        bfr[j].h[1] = *(const v8bf*)(p + 16);
      }
#pragma unroll
      for (int i = 0; i < 2; ++i)
#pragma unroll
        for (int j = 0; j < 4; ++j)
          acc[i][j] = __builtin_amdgcn_wmma_f32_16x16x32_bf16(
              false, afr[i].v, false, bfr[j].v, (short)0, acc[i][j], false, false);
    }

    // stage next buffer
    *(uint4*)(sA0 + nxt * bufStep)             = a0;
    *(uint4*)(sA0 + nxt * bufStep + 64 * LSTR) = a1;
    *(uint4*)(sB0 + nxt * bufStep)             = b0;
    *(uint4*)(sB0 + nxt * bufStep + 64 * LSTR) = b1;
    __syncthreads();
  }

  // epilogue: compute last stage
  {
    const int cur = (nStages - 1) & 1;
    union F { v16bf v; v8bf h[2]; };
    F afr[2], bfr[4];
#pragma unroll
    for (int i = 0; i < 2; ++i) {
      const unsigned short* p = &sA[cur][(waveM + 16 * i + lr) * LSTR];
      afr[i].h[0] = *(const v8bf*)(p + aKhalf);
      afr[i].h[1] = *(const v8bf*)(p + aKhalf + 16);
    }
#pragma unroll
    for (int j = 0; j < 4; ++j) {
      const unsigned short* p = &sB[cur][(waveN + 16 * j + lr) * LSTR + bK0];
      bfr[j].h[0] = *(const v8bf*)(p);
      bfr[j].h[1] = *(const v8bf*)(p + 16);
    }
#pragma unroll
    for (int i = 0; i < 2; ++i)
#pragma unroll
      for (int j = 0; j < 4; ++j)
        acc[i][j] = __builtin_amdgcn_wmma_f32_16x16x32_bf16(
            false, afr[i].v, false, bfr[j].v, (short)0, acc[i][j], false, false);
  }

  // C/D layout: vgpr v -> M = v (lanes 0..15) or v+8 (lanes 16..31); N = lane%16
  const int crow = rowBlk + waveM + ((lane >> 4) * 8);
  const int ccol = colBlk + waveN + lr;
#pragma unroll
  for (int i = 0; i < 2; ++i)
#pragma unroll
    for (int j = 0; j < 4; ++j)
#pragma unroll
      for (int v = 0; v < 8; ++v)
        C[(size_t)(crow + 16 * i + v) * N + ccol + 16 * j] = acc[i][j][v];
}

// ---------------- column stats pass 1: partial sums over row slabs ----------
// grid (N/64, SPART); 256 thr = 64 cols x 4 row sub-partitions
__global__ __launch_bounds__(256) void colstats_part(const float* __restrict__ Y,
                                                     float* __restrict__ psum,
                                                     float* __restrict__ psq,
                                                     int rows, int N) {
  const int col  = blockIdx.x * 64 + (threadIdx.x & 63);
  const int part = threadIdx.x >> 6;
  const int slab = rows / SPART;
  const int rbeg = blockIdx.y * slab;
  float s = 0.f, q = 0.f;
  for (int r = rbeg + part; r < rbeg + slab; r += 4) {
    float v = Y[(size_t)r * N + col];
    s += v;
    q += v * v;
  }
  __shared__ float ss[256], sq[256];
  ss[threadIdx.x] = s;
  sq[threadIdx.x] = q;
  __syncthreads();
  if (part == 0) {
#pragma unroll
    for (int p = 1; p < 4; ++p) {
      s += ss[p * 64 + (threadIdx.x & 63)];
      q += sq[p * 64 + (threadIdx.x & 63)];
    }
    psum[(size_t)blockIdx.y * N + col] = s;
    psq[(size_t)blockIdx.y * N + col]  = q;
  }
}

// ---------------- column stats pass 2: reduce partials -> mean/rstd ----------
__global__ __launch_bounds__(256) void colstats_final(const float* __restrict__ psum,
                                                      const float* __restrict__ psq,
                                                      float* __restrict__ mean,
                                                      float* __restrict__ rstd,
                                                      int rows, int N) {
  const int col = blockIdx.x * 256 + threadIdx.x;
  if (col >= N) return;
  float s = 0.f, q = 0.f;
  for (int p = 0; p < SPART; ++p) {
    s += psum[(size_t)p * N + col];
    q += psq[(size_t)p * N + col];
  }
  float m   = s / (float)rows;
  float var = q / (float)rows - m * m;
  mean[col] = m;
  rstd[col] = rsqrtf(var + EPSV);
}

// ---------------- BN + ReLU epilogue (N must be power of two) ----------------
__global__ __launch_bounds__(256) void bn_relu_kernel(const float* __restrict__ Y,
                                                      const float* __restrict__ mean,
                                                      const float* __restrict__ rstd,
                                                      const float* __restrict__ g,
                                                      const float* __restrict__ be,
                                                      unsigned short* __restrict__ out_bf,
                                                      float* __restrict__ out_f,
                                                      int N, size_t total) {
  size_t i = (size_t)blockIdx.x * blockDim.x + threadIdx.x;
  size_t stride = (size_t)gridDim.x * blockDim.x;
  for (; i < total; i += stride) {
    int col = (int)(i & (size_t)(N - 1));
    float v = (Y[i] - mean[col]) * rstd[col] * g[col] + be[col];
    v = fmaxf(v, 0.f);
    if (out_bf) out_bf[i] = f2bf(v);
    if (out_f)  out_f[i]  = v;
  }
}

// ---------------- logits: wave-per-row GEMV with K=11 outputs ----------------
__global__ __launch_bounds__(256) void logits_kernel(const float* __restrict__ H,
                                                     const float* __restrict__ W,   // [H2,K]
                                                     const float* __restrict__ bias,
                                                     float* __restrict__ out) {     // [B,K]
  const int row  = (int)((blockIdx.x * (size_t)blockDim.x + threadIdx.x) >> 5);
  const int lane = threadIdx.x & 31;
  if (row >= BB) return;
  const float* hrow = H + (size_t)row * HH2;
  float acc[KK];
#pragma unroll
  for (int k = 0; k < KK; ++k) acc[k] = 0.f;
  for (int i = lane; i < HH2; i += 32) {
    float h = hrow[i];
    const float* wrow = W + (size_t)i * KK;
#pragma unroll
    for (int k = 0; k < KK; ++k) acc[k] += h * wrow[k];
  }
#pragma unroll
  for (int k = 0; k < KK; ++k)
    for (int off = 16; off; off >>= 1) acc[k] += __shfl_xor(acc[k], off, 32);
  if (lane == 0) {
#pragma unroll
    for (int k = 0; k < KK; ++k) out[(size_t)row * KK + k] = acc[k] + bias[k];
  }
}

// ---------------- head layer 3: 512-dot + sigmoid into pcts[:,k] ----------------
__global__ __launch_bounds__(256) void head3_kernel(const float* __restrict__ R2,  // [B,512]
                                                    const float* __restrict__ w,   // [512]
                                                    const float* __restrict__ b3,  // [K]
                                                    int k,
                                                    float* __restrict__ pcts) {    // [B,K]
  const int row  = (int)((blockIdx.x * (size_t)blockDim.x + threadIdx.x) >> 5);
  const int lane = threadIdx.x & 31;
  if (row >= BB) return;
  const float* r = R2 + (size_t)row * DD;
  float s = 0.f;
  for (int i = lane; i < DD; i += 32) s += r[i] * w[i];
  for (int off = 16; off; off >>= 1) s += __shfl_xor(s, off, 32);
  if (lane == 0) {
    float z = s + b3[k];
    pcts[(size_t)row * KK + k] = 1.f / (1.f + __expf(-z));
  }
}

// ---------------- finalize: softmax, threshold/top3, 10^x, argmax ----------------
__global__ __launch_bounds__(256) void finalize_kernel(const float* __restrict__ logits,  // [B,K]
                                                       const float* __restrict__ pcts,    // [B,K]
                                                       float* __restrict__ final_preds,   // [B]
                                                       float* __restrict__ atp_max) {     // [B]
  const int b = blockIdx.x * blockDim.x + threadIdx.x;
  if (b >= BB) return;
  float lg[KK], p[KK], atp[KK];
  float mx = -3.4e38f;
#pragma unroll
  for (int k = 0; k < KK; ++k) { lg[k] = logits[(size_t)b * KK + k]; mx = fmaxf(mx, lg[k]); }
  float se = 0.f;
#pragma unroll
  for (int k = 0; k < KK; ++k) { p[k] = __expf(lg[k] - mx); se += p[k]; }
  float inv = 1.f / se;
#pragma unroll
  for (int k = 0; k < KK; ++k) p[k] *= inv;
#pragma unroll
  for (int k = 0; k < KK; ++k) {
    float t = ((float)k - 6.0f) + pcts[(size_t)b * KK + k];
    atp[k] = __expf(2.302585092994046f * t);  // 10^t
  }
  bool any = false;
#pragma unroll
  for (int k = 0; k < KK; ++k) any = any || (p[k] > THRV);
  float wsum = 0.f, fsum = 0.f;
  if (any) {
#pragma unroll
    for (int k = 0; k < KK; ++k)
      if (p[k] > THRV) { wsum += p[k]; fsum += p[k] * atp[k]; }
  } else {
    int t1 = 0, t2 = 0, t3 = 0;
    float v1 = -1.f, v2 = -1.f, v3 = -1.f;
#pragma unroll
    for (int k = 0; k < KK; ++k) {
      float v = p[k];
      if (v > v1)      { v3 = v2; t3 = t2; v2 = v1; t2 = t1; v1 = v; t1 = k; }
      else if (v > v2) { v3 = v2; t3 = t2; v2 = v; t2 = k; }
      else if (v > v3) { v3 = v; t3 = k; }
    }
    wsum = v1 + v2 + v3;
    fsum = v1 * atp[t1] + v2 * atp[t2] + v3 * atp[t3];
  }
  final_preds[b] = fsum / wsum;
  int mb = 0; float bv = p[0];
#pragma unroll
  for (int k = 1; k < KK; ++k) if (p[k] > bv) { bv = p[k]; mb = k; }
  atp_max[b] = atp[mb];
}

// ---------------- host launcher ----------------
extern "C" void kernel_launch(void* const* d_in, const int* in_sizes, int n_in,
                              void* d_out, int out_size, void* d_ws, size_t ws_size,
                              hipStream_t stream) {
  (void)in_sizes; (void)n_in; (void)out_size; (void)ws_size;
  const float* x    = (const float*)d_in[0];
  const float* Wc1  = (const float*)d_in[1];
  // bc1 (d_in[2]) cancels under BN
  const float* gc1  = (const float*)d_in[3];
  const float* bec1 = (const float*)d_in[4];
  const float* Wc2  = (const float*)d_in[5];
  // bc2 (d_in[6]) cancels under BN
  const float* gc2  = (const float*)d_in[7];
  const float* bec2 = (const float*)d_in[8];
  const float* Wc3  = (const float*)d_in[9];
  const float* bc3  = (const float*)d_in[10];
  const float* Rw1  = (const float*)d_in[11];
  // Rb1 (d_in[12]) cancels under BN
  const float* Rg1  = (const float*)d_in[13];
  const float* Rbe1 = (const float*)d_in[14];
  const float* Rw2  = (const float*)d_in[15];
  // Rb2 (d_in[16]) cancels under BN
  const float* Rg2  = (const float*)d_in[17];
  const float* Rbe2 = (const float*)d_in[18];
  const float* Rw3  = (const float*)d_in[19];
  const float* Rb3  = (const float*)d_in[20];

  // output regions: final[B], logits[B*K], pcts[B*K], atp_max[B]
  float* out_final  = (float*)d_out;
  float* out_logits = out_final + BB;
  float* out_pcts   = out_logits + (size_t)BB * KK;
  float* out_atp    = out_pcts + (size_t)BB * KK;

  // workspace layout
  char* ws = (char*)d_ws;
  size_t off = 0;
  auto alloc = [&](size_t bytes) -> void* {
    void* p = ws + off;
    off += (bytes + 255) & ~(size_t)255;
    return p;
  };
  unsigned short* XB   = (unsigned short*)alloc((size_t)BB * DD * 2);         // x bf16 [B][D]
  unsigned short* WC1T = (unsigned short*)alloc((size_t)DD * HH1 * 2);        // [H1][D]
  unsigned short* WC2T = (unsigned short*)alloc((size_t)HH1 * HH2 * 2);       // [H2][H1]
  unsigned short* RW1T = (unsigned short*)alloc((size_t)KK * DD * HH1 * 2);   // [K][H1][D]
  unsigned short* RW2T = (unsigned short*)alloc((size_t)KK * HH1 * DD * 2);   // [K][D][H1]
  float* BUF1 = (float*)alloc((size_t)BB * HH2 * 4);  // h2 / per-head r2
  float* BUF2 = (float*)alloc((size_t)BB * HH1 * 4);  // h1 / per-head r1
  unsigned short* BB1 = (unsigned short*)alloc((size_t)BB * HH1 * 2);  // bf16 activations
  float* MEAN = (float*)alloc(HH2 * 4);
  float* RSTD = (float*)alloc(HH2 * 4);
  float* PSUM = (float*)alloc((size_t)SPART * HH2 * 4);
  float* PSQ  = (float*)alloc((size_t)SPART * HH2 * 4);

  // ---- conversions: x plain; weights converted + transposed (once) ----
  cvt_bf16_kernel<<<2048, 256, 0, stream>>>(x, XB, (size_t)BB * DD);
  cvt_t_kernel<<<dim3(HH1 / 32, DD / 32, 1), 256, 0, stream>>>(Wc1, WC1T, DD, HH1, (size_t)DD * HH1);
  cvt_t_kernel<<<dim3(HH2 / 32, HH1 / 32, 1), 256, 0, stream>>>(Wc2, WC2T, HH1, HH2, (size_t)HH1 * HH2);
  cvt_t_kernel<<<dim3(HH1 / 32, DD / 32, KK), 256, 0, stream>>>(Rw1, RW1T, DD, HH1, (size_t)DD * HH1);
  cvt_t_kernel<<<dim3(DD / 32, HH1 / 32, KK), 256, 0, stream>>>(Rw2, RW2T, HH1, DD, (size_t)HH1 * DD);

  const int ew_blocks = 4096;
  auto stats = [&](const float* Y, int N) {
    colstats_part<<<dim3(N / 64, SPART), 256, 0, stream>>>(Y, PSUM, PSQ, BB, N);
    colstats_final<<<(N + 255) / 256, 256, 0, stream>>>(PSUM, PSQ, MEAN, RSTD, BB, N);
  };

  // ---- classifier layer 1: h1 = BNReLU(x @ Wc1) ----
  gemm_bf16_wmma<<<dim3(HH1 / BN, BB / BM), 256, 0, stream>>>(XB, WC1T, BUF2, BB, HH1, DD);
  stats(BUF2, HH1);
  bn_relu_kernel<<<ew_blocks, 256, 0, stream>>>(BUF2, MEAN, RSTD, gc1, bec1,
                                                BB1, nullptr, HH1, (size_t)BB * HH1);

  // ---- classifier layer 2: h2 = BNReLU(h1 @ Wc2) ----
  gemm_bf16_wmma<<<dim3(HH2 / BN, BB / BM), 256, 0, stream>>>(BB1, WC2T, BUF1, BB, HH2, HH1);
  stats(BUF1, HH2);
  bn_relu_kernel<<<ew_blocks, 256, 0, stream>>>(BUF1, MEAN, RSTD, gc2, bec2,
                                                nullptr, BUF1, HH2, (size_t)BB * HH2);

  // ---- logits (before heads reuse BUF1) ----
  logits_kernel<<<(BB * 32) / 256, 256, 0, stream>>>(BUF1, Wc3, bc3, out_logits);

  // ---- K regression heads ----
  for (int k = 0; k < KK; ++k) {
    const unsigned short* w1 = RW1T + (size_t)k * DD * HH1;
    const unsigned short* w2 = RW2T + (size_t)k * HH1 * DD;
    gemm_bf16_wmma<<<dim3(HH1 / BN, BB / BM), 256, 0, stream>>>(XB, w1, BUF2, BB, HH1, DD);
    stats(BUF2, HH1);
    bn_relu_kernel<<<ew_blocks, 256, 0, stream>>>(BUF2, MEAN, RSTD,
                                                  Rg1 + (size_t)k * HH1, Rbe1 + (size_t)k * HH1,
                                                  BB1, nullptr, HH1, (size_t)BB * HH1);
    gemm_bf16_wmma<<<dim3(DD / BN, BB / BM), 256, 0, stream>>>(BB1, w2, BUF1, BB, DD, HH1);
    stats(BUF1, DD);
    bn_relu_kernel<<<ew_blocks, 256, 0, stream>>>(BUF1, MEAN, RSTD,
                                                  Rg2 + (size_t)k * DD, Rbe2 + (size_t)k * DD,
                                                  nullptr, BUF1, DD, (size_t)BB * DD);
    head3_kernel<<<(BB * 32) / 256, 256, 0, stream>>>(BUF1, Rw3 + (size_t)k * DD, Rb3, k, out_pcts);
  }

  // ---- finalize ----
  finalize_kernel<<<BB / 256, 256, 0, stream>>>(out_logits, out_pcts, out_final, out_atp);
}
